// SelfAttentionMasked_51075751084165
// MI455X (gfx1250) — matrix operand order repaired and checked by reading, not compiled, not run on hardware
//
#include <hip/hip_runtime.h>
#include <math.h>

// ---------------------------------------------------------------------------
// MI455X (gfx1250) masked self-attention, bf16 WMMA pipeline.
// wave32; WMMA 16x16x32 bf16 -> f32.
// ---------------------------------------------------------------------------

typedef __attribute__((ext_vector_type(16))) __bf16 bf16x16_t;
typedef __attribute__((ext_vector_type(8)))  __bf16 bf16x8_t;
typedef __attribute__((ext_vector_type(8)))  float  f32x8_t;

#define N_SEQ   1024
#define Q_DIM   640
#define N_HEADS 8
#define D_HEAD  80
#define D_PAD   96          // head dim padded to 3 * 32 for the WMMA K loop
#define BATCH   2
#define BH      (BATCH * N_HEADS)          // 16
#define M_ROWS  (BATCH * N_SEQ)            // 2048 rows into projections
#define OUT_ROWS (2 * BATCH * N_SEQ)       // 4096 rows of final output

__device__ __forceinline__ f32x8_t wmma_bf16(bf16x16_t a, bf16x16_t b, f32x8_t c) {
  return __builtin_amdgcn_wmma_f32_16x16x32_bf16(false, a, false, b, (short)0, c,
                                                 false, false);
}

__device__ __forceinline__ f32x8_t zero8() {
  f32x8_t z;
#pragma unroll
  for (int i = 0; i < 8; ++i) z[i] = 0.0f;
  return z;
}

// A-fragment / B-fragment builders (16-bit data, contiguous k = 16*hi + s, same order in A and B).
// Both need two contiguous 8-element (16-byte) loads per lane.
__device__ __forceinline__ bf16x16_t frag_from2(const __bf16* p0, const __bf16* p1) {
  bf16x8_t a = *(const bf16x8_t*)p0;
  bf16x8_t b = *(const bf16x8_t*)p1;
  bf16x16_t f;
#pragma unroll
  for (int i = 0; i < 8; ++i) { f[i] = a[i]; f[i + 8] = b[i]; }
  return f;
}
__device__ __forceinline__ bf16x16_t frag16(const __bf16* p) {
  return frag_from2(p, p + 8);
}

// ---------------------------------------------------------------------------
// Prep kernels
// ---------------------------------------------------------------------------
__global__ void cast_f32_bf16(const float* __restrict__ s, __bf16* __restrict__ d, int n) {
  int i = blockIdx.x * blockDim.x + threadIdx.x;
  if (i < n) d[i] = (__bf16)s[i];
}

__global__ void zero_bf16(__bf16* __restrict__ d, int n) {
  int i = blockIdx.x * blockDim.x + threadIdx.x;
  if (i < n) d[i] = (__bf16)0.0f;
}

// W[k][j] (640x640 row-major) -> Wt[j][k] bf16 (column-contiguous K for B frags)
__global__ void transpose_cast(const float* __restrict__ W, __bf16* __restrict__ Wt) {
  int i = blockIdx.x * blockDim.x + threadIdx.x;
  if (i < Q_DIM * Q_DIM) {
    int k = i / Q_DIM, j = i % Q_DIM;
    Wt[(size_t)j * Q_DIM + k] = (__bf16)W[(size_t)k * Q_DIM + j];
  }
}

// ---------------------------------------------------------------------------
// QKV projection: one 16x16 tile per wave.  X[2048][640] @ Wt^T -> 2048x640.
// mode 0: store split-head row-major q/k  [bh][n][96] (cols 80..95 pre-zeroed)
// mode 1: store split-head transposed v   [bh][c][n]
// ---------------------------------------------------------------------------
__global__ void proj_gemm(const __bf16* __restrict__ X, const __bf16* __restrict__ Wt,
                          __bf16* __restrict__ outQK, __bf16* __restrict__ outVT,
                          int mode) {
  const int lane = threadIdx.x & 31;
  const int l = lane & 15;
  const int hi = lane >> 4;
  const int w = blockIdx.x * 8 + (threadIdx.x >> 5);
  const int tm = w / 40;             // 128 row tiles
  const int tn = w % 40;             // 40 col tiles

  const __bf16* arow = X + ((size_t)tm * 16 + l) * Q_DIM;
  const __bf16* bcol = Wt + ((size_t)tn * 16 + l) * Q_DIM + hi * 16;

  f32x8_t acc = zero8();
#pragma unroll 4
  for (int kk = 0; kk < Q_DIM; kk += 32) {
    bf16x16_t a = frag_from2(arow + kk + hi * 16, arow + kk + hi * 16 + 8);
    bf16x16_t b = frag16(bcol + kk);
    acc = wmma_bf16(a, b, acc);
  }

  const int j = tn * 16 + l;         // global output column
  const int h = j / D_HEAD, c = j % D_HEAD;
#pragma unroll
  for (int r = 0; r < 8; ++r) {
    const int row = tm * 16 + r + hi * 8;     // 0..2047
    const int b_ = row >> 10, n = row & (N_SEQ - 1);
    const __bf16 v = (__bf16)acc[r];
    if (mode == 0)
      outQK[(((size_t)(b_ * N_HEADS + h) * N_SEQ) + n) * D_PAD + c] = v;
    else
      outVT[(((size_t)(b_ * N_HEADS + h) * D_HEAD) + c) * N_SEQ + n] = v;
  }
}

// ---------------------------------------------------------------------------
// Fused fg+bg flash attention.  One wave per (bh, qtile-of-16); S tiles are
// computed once and masked both ways so the 64MB mask is streamed exactly once.
// ---------------------------------------------------------------------------
__global__ void attn_kernel(const __bf16* __restrict__ qb, const __bf16* __restrict__ kb,
                            const __bf16* __restrict__ vt, const float* __restrict__ mask,
                            __bf16* __restrict__ ao) {
  const int lane  = threadIdx.x & 31;
  const int l     = lane & 15;
  const int hi    = lane >> 4;
  const int wslot = threadIdx.x >> 5;          // 4 waves / block
  const int w  = blockIdx.x * 4 + wslot;       // 1024 waves total
  const int qt = w & 63;                       // 64 query tiles
  const int bh = w >> 6;                       // 0..15

  __shared__ __align__(16) __bf16 plds[4][16][32];

  const float scale = 0.1118033988749895f;     // 80^-0.5
  const float NEG   = -3.4028234663852886e38f; // finfo(float32).min

  // Q fragments: 16x96 bf16, 3 K-chunks, held for the whole key loop.
  bf16x16_t qf[3];
  const __bf16* qrow = qb + ((size_t)bh * N_SEQ + qt * 16 + l) * D_PAD;
#pragma unroll
  for (int c = 0; c < 3; ++c)
    qf[c] = frag_from2(qrow + c * 32 + hi * 16, qrow + c * 32 + hi * 16 + 8);

  f32x8_t o[2][5];
  float rmax[2][8], rsum[2][8];
#pragma unroll
  for (int fb = 0; fb < 2; ++fb) {
#pragma unroll
    for (int t = 0; t < 5; ++t) o[fb][t] = zero8();
#pragma unroll
    for (int r = 0; r < 8; ++r) { rmax[fb][r] = -INFINITY; rsum[fb][r] = 0.0f; }
  }

  const float* mbase = mask + ((size_t)bh << 20);   // bh * 1024 * 1024

  for (int kc = 0; kc < N_SEQ / 32; ++kc) {
    // --- S = Q K^T for a 16x32 key chunk (2 N-tiles x 3 K-chunks) ---
    f32x8_t s0 = zero8(), s1 = zero8();
    const __bf16* k0 = kb + ((size_t)bh * N_SEQ + kc * 32 + l) * D_PAD + hi * 16;
    const __bf16* k1 = kb + ((size_t)bh * N_SEQ + kc * 32 + 16 + l) * D_PAD + hi * 16;
#pragma unroll
    for (int c = 0; c < 3; ++c) {
      s0 = wmma_bf16(qf[c], frag16(k0 + c * 32), s0);
      s1 = wmma_bf16(qf[c], frag16(k1 + c * 32), s1);
    }

    // --- scale + mask load (single pass over box_mask) ---
    float sm0[8], sm1[8], mv0[8], mv1[8];
#pragma unroll
    for (int r = 0; r < 8; ++r) {
      const int qr = qt * 16 + r + hi * 8;
      mv0[r] = mbase[(size_t)qr * N_SEQ + kc * 32 + l];
      mv1[r] = mbase[(size_t)qr * N_SEQ + kc * 32 + 16 + l];
      sm0[r] = s0[r] * scale;
      sm1[r] = s1[r] * scale;
    }

#pragma unroll
    for (int fb = 0; fb < 2; ++fb) {
      const float addv = (fb == 0) ? 1.0f : 0.0f;
      float p0[8], p1[8];
#pragma unroll
      for (int r = 0; r < 8; ++r) {
        const bool k0keep = (mv0[r] != 0.0f) == (fb == 0);
        const bool k1keep = (mv1[r] != 0.0f) == (fb == 0);
        p0[r] = k0keep ? (sm0[r] + addv) : NEG;
        p1[r] = k1keep ? (sm1[r] + addv) : NEG;
      }
      // --- online softmax, row stats across the 16-lane half-groups ---
#pragma unroll
      for (int r = 0; r < 8; ++r) {
        float m = fmaxf(p0[r], p1[r]);
        for (int sh = 1; sh < 16; sh <<= 1) m = fmaxf(m, __shfl_xor(m, sh, 32));
        const float nm    = fmaxf(rmax[fb][r], m);
        const float alpha = __expf(rmax[fb][r] - nm);
        rmax[fb][r] = nm;
        const float e0 = __expf(p0[r] - nm);
        const float e1 = __expf(p1[r] - nm);
        float sl = e0 + e1;
        for (int sh = 1; sh < 16; sh <<= 1) sl += __shfl_xor(sl, sh, 32);
        rsum[fb][r] = rsum[fb][r] * alpha + sl;
        p0[r] = e0; p1[r] = e1;
#pragma unroll
        for (int t = 0; t < 5; ++t) o[fb][t][r] *= alpha;
      }
      // --- C-layout -> A-layout for P via LDS (1KB per wave) ---
#pragma unroll
      for (int r = 0; r < 8; ++r) {
        plds[wslot][r + hi * 8][l]      = (__bf16)p0[r];
        plds[wslot][r + hi * 8][16 + l] = (__bf16)p1[r];
      }
      // cross-lane LDS dependency: drain DS counter explicitly (CDNA5 split cnt)
      asm volatile("s_wait_dscnt 0" ::: "memory");
      bf16x16_t pf = frag_from2(&plds[wslot][l][hi * 16], &plds[wslot][l][hi * 16 + 8]);
      // --- O += P @ V (V stored [bh][c][key]: contiguous B fragments) ---
#pragma unroll
      for (int t = 0; t < 5; ++t) {
        const __bf16* vp =
            vt + ((size_t)bh * D_HEAD + t * 16 + l) * N_SEQ + kc * 32 + hi * 16;
        o[fb][t] = wmma_bf16(pf, frag16(vp), o[fb][t]);
      }
    }
  }

  // --- normalize + store [fb*2+b][n][h*80+c] bf16 ---
  const int b = bh >> 3, h = bh & 7;
#pragma unroll
  for (int fb = 0; fb < 2; ++fb) {
    const size_t obase = (size_t)(fb * BATCH + b) * N_SEQ * Q_DIM;
#pragma unroll
    for (int t = 0; t < 5; ++t) {
#pragma unroll
      for (int r = 0; r < 8; ++r) {
        const int n = qt * 16 + r + hi * 8;
        const int c = h * D_HEAD + t * 16 + l;
        ao[obase + (size_t)n * Q_DIM + c] = (__bf16)(o[fb][t][r] / rsum[fb][r]);
      }
    }
  }
}

// ---------------------------------------------------------------------------
// Output projection: [4096][640] bf16 @ Wo + bo -> f32
// ---------------------------------------------------------------------------
__global__ void out_gemm(const __bf16* __restrict__ A, const __bf16* __restrict__ Wt,
                         const float* __restrict__ bias, float* __restrict__ out) {
  const int lane = threadIdx.x & 31;
  const int l = lane & 15;
  const int hi = lane >> 4;
  const int w = blockIdx.x * 8 + (threadIdx.x >> 5);
  const int tm = w / 40;             // 256 row tiles
  const int tn = w % 40;

  const __bf16* arow = A + ((size_t)tm * 16 + l) * Q_DIM;
  const __bf16* bcol = Wt + ((size_t)tn * 16 + l) * Q_DIM + hi * 16;

  f32x8_t acc = zero8();
#pragma unroll 4
  for (int kk = 0; kk < Q_DIM; kk += 32) {
    bf16x16_t a = frag_from2(arow + kk + hi * 16, arow + kk + hi * 16 + 8);
    bf16x16_t b = frag16(bcol + kk);
    acc = wmma_bf16(a, b, acc);
  }

  const int j = tn * 16 + l;
  const float bj = bias[j];
#pragma unroll
  for (int r = 0; r < 8; ++r) {
    const int row = tm * 16 + r + hi * 8;
    out[(size_t)row * Q_DIM + j] = acc[r] + bj;
  }
}

// ---------------------------------------------------------------------------
extern "C" void kernel_launch(void* const* d_in, const int* in_sizes, int n_in,
                              void* d_out, int out_size, void* d_ws, size_t ws_size,
                              hipStream_t stream) {
  (void)in_sizes; (void)n_in; (void)out_size; (void)ws_size;
  const float* x    = (const float*)d_in[0];
  const float* mask = (const float*)d_in[1];
  const float* Wq   = (const float*)d_in[2];
  const float* Wk   = (const float*)d_in[3];
  const float* Wv   = (const float*)d_in[4];
  const float* Wo   = (const float*)d_in[5];
  const float* bo   = (const float*)d_in[6];
  float* out = (float*)d_out;

  char* ws = (char*)d_ws;
  size_t off = 0;
  auto carve = [&](size_t elems) -> __bf16* {
    off = (off + 255) & ~(size_t)255;
    __bf16* p = (__bf16*)(ws + off);
    off += elems * sizeof(__bf16);
    return p;
  };

  const int NX = M_ROWS * Q_DIM;                 // 2048*640
  const int NW = Q_DIM * Q_DIM;                  // 640*640
  const int NQK = BH * N_SEQ * D_PAD;            // 16*1024*96
  const int NVT = BH * D_HEAD * N_SEQ;           // 16*80*1024
  const int NAO = OUT_ROWS * Q_DIM;              // 4096*640

  __bf16* xw  = carve(NX);
  __bf16* wtq = carve(NW);
  __bf16* wtk = carve(NW);
  __bf16* wtv = carve(NW);
  __bf16* wto = carve(NW);
  __bf16* qb  = carve(NQK);
  __bf16* kb  = carve(NQK);
  __bf16* vt  = carve(NVT);
  __bf16* ao  = carve(NAO);

  cast_f32_bf16<<<(NX + 255) / 256, 256, 0, stream>>>(x, xw, NX);
  transpose_cast<<<(NW + 255) / 256, 256, 0, stream>>>(Wq, wtq);
  transpose_cast<<<(NW + 255) / 256, 256, 0, stream>>>(Wk, wtk);
  transpose_cast<<<(NW + 255) / 256, 256, 0, stream>>>(Wv, wtv);
  transpose_cast<<<(NW + 255) / 256, 256, 0, stream>>>(Wo, wto);
  zero_bf16<<<(NQK + 255) / 256, 256, 0, stream>>>(qb, NQK);
  zero_bf16<<<(NQK + 255) / 256, 256, 0, stream>>>(kb, NQK);

  // 128*40 tiles / 8 waves per block = 640 blocks
  proj_gemm<<<640, 256, 0, stream>>>(xw, wtq, qb, nullptr, 0);
  proj_gemm<<<640, 256, 0, stream>>>(xw, wtk, kb, nullptr, 0);
  proj_gemm<<<640, 256, 0, stream>>>(xw, wtv, nullptr, vt, 1);

  // 16 bh * 64 qtiles = 1024 waves / 4 per block = 256 blocks
  attn_kernel<<<256, 128, 0, stream>>>(qb, kb, vt, mask, ao);

  // 256*40 tiles / 8 waves per block = 1280 blocks
  out_gemm<<<1280, 256, 0, stream>>>(ao, wto, bo, out);
}
